// Aligner_20358144983476
// MI455X (gfx1250) — matrix operand order repaired and checked by reading, not compiled, not run on hardware
//
#include <hip/hip_runtime.h>
#include <hip/hip_bf16.h>

typedef __bf16 bf16_t;
typedef __attribute__((ext_vector_type(16))) __bf16 v16bf;
typedef __attribute__((ext_vector_type(8)))  __bf16 v8bf;
typedef __attribute__((ext_vector_type(8)))  float   v8f;

// Problem constants (match reference)
#define CB     64
#define CT     512
#define CD     512
#define CRNN   1024
#define CATT   128
#define COUT   80
#define CSPK   64
#define CNF    32
#define CKS    31
#define CTOUT  600
#define CKCAT  1600   // D + SPK + RNN = 512+64+1024

// ---------------------------------------------------------------------------
// Generic bf16 WMMA GEMM:  C[M x N] (f32) = A[M x K] (bf16, row-major, lda)
//                                        x B^T, where Bm is [N x K] (bf16, ldb)
// One wave computes one 16x16 C tile. K must be a multiple of 32.
// ---------------------------------------------------------------------------
__global__ void gemm_bf16_wmma(const bf16_t* __restrict__ A, int lda,
                               const bf16_t* __restrict__ Bm, int ldb,
                               float* __restrict__ C, int ldc,
                               int tilesN, int totalTiles, int K)
{
    int wave = (int)((blockIdx.x * blockDim.x + threadIdx.x) >> 5);
    if (wave >= totalTiles) return;                 // wave-uniform exit
    int tm = wave / tilesN;
    int tn = wave - tm * tilesN;
    int lane = threadIdx.x & 31;
    int lh = lane >> 4;
    int ll = lane & 15;

    const bf16_t* arow = A  + (size_t)(tm * 16 + ll) * (size_t)lda;
    const bf16_t* brow = Bm + (size_t)(tn * 16 + ll) * (size_t)ldb;

    v8f acc = {};
    for (int k = 0; k < K; k += 32) {
        union { v16bf v; v8bf h[2]; } a, b;
        a.h[0] = *(const v8bf*)(arow + k + lh * 16);
        a.h[1] = *(const v8bf*)(arow + k + lh * 16 + 8);
        b.v    = *(const v16bf*)(brow + k + lh * 16);
        acc = __builtin_amdgcn_wmma_f32_16x16x32_bf16(
                  false, a.v, false, b.v, (short)0, acc, false, false);
    }

    float* cbase = C + (size_t)(tm * 16 + lh * 8) * (size_t)ldc + tn * 16 + ll;
    #pragma unroll
    for (int r = 0; r < 8; ++r)
        cbase[(size_t)r * (size_t)ldc] = acc[r];
}

// ---------------------------------------------------------------------------
// Fused gates GEMM (M=64, N=4096 permuted (unit,gate), K=1600) + LSTM cell.
// Each wave: 1 M-tile x 2 N-tiles (A reused across 2 wmma per k-step).
// Epilogue: spill C tiles to this wave's LDS slot, then each lane handles 4
// (batch,unit) pairs: read 4 gates from LDS, add biases, LSTM update, write
// c (f32), h (f32) and h (bf16 into NEXT xcat buffer).
// Column permutation: col n' = unit*4 + gate  (orig row = gate*1024 + unit).
// ---------------------------------------------------------------------------
__global__ void gates_lstm_wmma(const bf16_t* __restrict__ A,     // xcat (cur)
                                const bf16_t* __restrict__ Bm,    // wcat perm
                                const float*  __restrict__ bsum,  // permuted biases
                                float* __restrict__ cbuf,
                                float* __restrict__ hbuf,
                                bf16_t* __restrict__ xnxt)        // xcat (next)
{
    const int NT2  = (4 * CRNN) / 32;        // 128 N-tile pairs
    const int JOBS = (CB / 16) * NT2;        // 512 wave jobs
    __shared__ float tileLds[8 * 512];       // 2 tiles of 16x16 per wave

    int wave = (int)((blockIdx.x * blockDim.x + threadIdx.x) >> 5);
    if (wave >= JOBS) return;                // wave-uniform
    int tm  = wave / NT2;
    int tj  = wave - tm * NT2;
    int tn0 = tj * 2;
    int lane = threadIdx.x & 31;
    int lh = lane >> 4;
    int ll = lane & 15;
    int wslot = ((threadIdx.x >> 5) & 7) * 512;

    const bf16_t* arow = A  + (size_t)(tm * 16 + ll) * CKCAT;
    const bf16_t* b0   = Bm + (size_t)(tn0 * 16 + ll) * CKCAT;
    const bf16_t* b1   = b0 + (size_t)16 * CKCAT;

    v8f acc0 = {}, acc1 = {};
    for (int k = 0; k < CKCAT; k += 32) {
        union { v16bf v; v8bf h[2]; } a, bb0, bb1;
        a.h[0] = *(const v8bf*)(arow + k + lh * 16);
        a.h[1] = *(const v8bf*)(arow + k + lh * 16 + 8);
        bb0.v  = *(const v16bf*)(b0 + k + lh * 16);
        bb1.v  = *(const v16bf*)(b1 + k + lh * 16);
        acc0 = __builtin_amdgcn_wmma_f32_16x16x32_bf16(
                   false, a.v, false, bb0.v, (short)0, acc0, false, false);
        acc1 = __builtin_amdgcn_wmma_f32_16x16x32_bf16(
                   false, a.v, false, bb1.v, (short)0, acc1, false, false);
    }

    // Spill both C tiles: tile layout [row(b) 0..15][col 0..15]
    #pragma unroll
    for (int r = 0; r < 8; ++r) {
        tileLds[wslot +       (lh * 8 + r) * 16 + ll] = acc0[r];
        tileLds[wslot + 256 + (lh * 8 + r) * 16 + ll] = acc1[r];
    }

    // LSTM epilogue: 2 tiles * 16 batch rows * 4 units = 128 pairs, 4/lane
    #pragma unroll
    for (int pp = 0; pp < 4; ++pp) {
        int p  = lane + pp * 32;
        int tt = p >> 6;                  // tile 0/1
        int q  = p & 63;
        int br = q & 15;                  // batch row within tile
        int ul = q >> 4;                  // unit within tile (0..3)
        int base = wslot + tt * 256 + br * 16 + 4 * ul;
        int colb = (tn0 + tt) * 16 + 4 * ul;     // permuted gate-i column
        int unit = colb >> 2;                    // global hidden unit
        float gi = tileLds[base + 0] + bsum[colb + 0];
        float gf = tileLds[base + 1] + bsum[colb + 1];
        float gg = tileLds[base + 2] + bsum[colb + 2];
        float go = tileLds[base + 3] + bsum[colb + 3];
        int bg = tm * 16 + br;
        size_t ci = (size_t)bg * CRNN + unit;
        float si = 1.f / (1.f + __expf(-gi));
        float sf = 1.f / (1.f + __expf(-gf));
        float so = 1.f / (1.f + __expf(-go));
        float cn = sf * cbuf[ci] + si * tanhf(gg);
        cbuf[ci] = cn;
        float hn = so * tanhf(cn);
        hbuf[ci] = hn;
        xnxt[(size_t)bg * CKCAT + (CD + CSPK) + unit] = (bf16_t)hn;
    }
}

// ---------------------------------------------------------------------------
// One-time conversions
// ---------------------------------------------------------------------------
__global__ void f32_to_bf16(const float* __restrict__ src,
                            bf16_t* __restrict__ dst, int n)
{
    int i = blockIdx.x * 256 + threadIdx.x;
    if (i < n) dst[i] = (bf16_t)src[i];
}

// Wcat permuted: row n' = unit*4+gate maps to orig row gate*1024+unit;
// cols 0..575 from W_ih, 576..1599 from W_hh.
__global__ void build_wcat(const float* __restrict__ wih,
                           const float* __restrict__ whh,
                           bf16_t* __restrict__ wcat)
{
    int i = blockIdx.x * 256 + threadIdx.x;
    if (i >= 4 * CRNN * CKCAT) return;
    int np = i / CKCAT, col = i - np * CKCAT;
    int unit = np >> 2, gate = np & 3;
    int orig = gate * CRNN + unit;
    float v = (col < (CD + CSPK)) ? wih[orig * (CD + CSPK) + col]
                                  : whh[orig * CRNN + (col - (CD + CSPK))];
    wcat[i] = (bf16_t)v;
}

__global__ void build_bsum(const float* __restrict__ b_ih,
                           const float* __restrict__ b_hh,
                           float* __restrict__ bsum)
{
    int n = blockIdx.x * 256 + threadIdx.x;
    if (n >= 4 * CRNN) return;
    int unit = n >> 2, gate = n & 3;
    int orig = gate * CRNN + unit;
    bsum[n] = b_ih[orig] + b_hh[orig];
}

// Initialize BOTH xcat buffers: [w_ctx(512)=0 | spk(64) | h(1024)=0]
__global__ void init_xcat(const int* __restrict__ spkr,
                          const float* __restrict__ spkr_emb,
                          bf16_t* __restrict__ xcat)
{
    int i = blockIdx.x * 256 + threadIdx.x;
    if (i >= 2 * CB * CKCAT) return;
    int col = i % CKCAT;
    int b = (i / CKCAT) % CB;
    float v = 0.f;
    if (col >= CD && col < CD + CSPK)
        v = spkr_emb[spkr[b] * CSPK + (col - CD)];
    xcat[i] = (bf16_t)v;
}

// ---------------------------------------------------------------------------
// Fused location-conv + energies.
// Block covers (b, 256 consecutive t). Halos of a_prev/a_cum, conv weights,
// loc_w and Wv staged in LDS (~28 KB). Each thread computes 32 conv filters
// then e[b,t] = sum_a Wv[a]*tanh(pq[b,a] + conv.loc_w[a,:] + pm[b,t,a]).
// ---------------------------------------------------------------------------
__global__ void energies_fused(const float* __restrict__ ap,
                               const float* __restrict__ ac,
                               const float* __restrict__ cw,
                               const float* __restrict__ loc_w,
                               const float* __restrict__ pq,
                               const float* __restrict__ pm,
                               const float* __restrict__ wv,
                               const unsigned char* __restrict__ mask,
                               float* __restrict__ e)
{
    __shared__ float sap[256 + CKS - 1];
    __shared__ float sac[256 + CKS - 1];
    __shared__ float scw[CNF * 2 * CKS];
    __shared__ float slw[CATT * CNF];
    __shared__ float swv[CATT];

    int b  = blockIdx.x >> 1;
    int t0 = (blockIdx.x & 1) * 256;
    int tid = threadIdx.x;

    for (int i = tid; i < 256 + CKS - 1; i += 256) {
        int t = t0 + i - (CKS - 1) / 2;
        bool ok = (t >= 0 && t < CT);
        sap[i] = ok ? ap[(size_t)b * CT + t] : 0.f;
        sac[i] = ok ? ac[(size_t)b * CT + t] : 0.f;
    }
    for (int i = tid; i < CNF * 2 * CKS; i += 256) scw[i] = cw[i];
    for (int i = tid; i < CATT * CNF; i += 256)    slw[i] = loc_w[i];
    if (tid < CATT) swv[tid] = wv[tid];
    __syncthreads();

    int idx = (size_t)b * CT + t0 + tid;

    float cf[CNF];
    #pragma unroll
    for (int f = 0; f < CNF; ++f) {
        const float* w0 = scw + f * (2 * CKS);
        const float* w1 = w0 + CKS;
        float acc = 0.f;
        #pragma unroll
        for (int k = 0; k < CKS; ++k)
            acc += sap[tid + k] * w0[k] + sac[tid + k] * w1[k];
        cf[f] = acc;
    }

    const float* pqb = pq + (size_t)b * CATT;
    const float* pmr = pm + (size_t)idx * CATT;
    float eacc = 0.f;
    for (int a = 0; a < CATT; ++a) {
        float loc = 0.f;
        #pragma unroll
        for (int f = 0; f < CNF; ++f) loc += cf[f] * slw[a * CNF + f];
        eacc += swv[a] * tanhf(pqb[a] + loc + pmr[a]);
    }
    e[idx] = mask[idx] ? eacc : -1e30f;
}

// ---------------------------------------------------------------------------
// Fused softmax + alignment update + context (bf16 memory) + projection.
// Block b; context held in LDS; projection: 80 outputs x 3 segments of 512
// mapped to 240 threads, combined via LDS partials.
// ---------------------------------------------------------------------------
__global__ void softmax_ctx_proj(const float* __restrict__ e,
                                 const bf16_t* __restrict__ memb,
                                 const float* __restrict__ hbuf,
                                 const float* __restrict__ Wp,
                                 const float* __restrict__ bp,
                                 float* __restrict__ aprev,
                                 float* __restrict__ acum,
                                 bf16_t* __restrict__ xnxt,
                                 float* __restrict__ out, int step)
{
    __shared__ float sw[CT];
    __shared__ float red[256];
    __shared__ float sctx[CD];
    __shared__ float part[COUT][3];

    int b = blockIdx.x;
    int tid = threadIdx.x;
    const float* eb = e + (size_t)b * CT;

    float m = -1e30f;
    for (int t = tid; t < CT; t += 256) m = fmaxf(m, eb[t]);
    red[tid] = m; __syncthreads();
    for (int s = 128; s > 0; s >>= 1) {
        if (tid < s) red[tid] = fmaxf(red[tid], red[tid + s]);
        __syncthreads();
    }
    m = red[0]; __syncthreads();

    float sum = 0.f;
    for (int t = tid; t < CT; t += 256) {
        float v = __expf(eb[t] - m);
        sw[t] = v; sum += v;
    }
    red[tid] = sum; __syncthreads();
    for (int s = 128; s > 0; s >>= 1) {
        if (tid < s) red[tid] += red[tid + s];
        __syncthreads();
    }
    float inv = 1.f / red[0]; __syncthreads();

    for (int t = tid; t < CT; t += 256) {
        float a = sw[t] * inv;
        sw[t] = a;
        aprev[(size_t)b * CT + t] = a;
        acum[(size_t)b * CT + t] += a;
    }
    __syncthreads();

    // context: w[d] = sum_t aw[t] * memory[b,t,d]  (bf16 memory, L2-resident)
    for (int d = tid; d < CD; d += 256) {
        float acc = 0.f;
        const bf16_t* mb = memb + (size_t)b * CT * CD + d;
        for (int t = 0; t < CT; ++t) acc += sw[t] * (float)mb[(size_t)t * CD];
        sctx[d] = acc;
        xnxt[(size_t)b * CKCAT + d] = (bf16_t)acc;
    }
    __syncthreads();

    // projection partials: o = tid/3, segment = tid%3 (h lo, h hi, context)
    if (tid < COUT * 3) {
        int o = tid / 3, seg = tid - o * 3;
        const float* wr = Wp + (size_t)o * (CRNN + CD);
        float acc = 0.f;
        if (seg < 2) {
            const float* hb = hbuf + (size_t)b * CRNN + seg * 512;
            const float* w  = wr + seg * 512;
            for (int x = 0; x < 512; ++x) acc += hb[x] * w[x];
        } else {
            for (int d = 0; d < CD; ++d) acc += sctx[d] * wr[CRNN + d];
        }
        part[o][seg] = acc;
    }
    __syncthreads();
    if (tid < COUT)
        out[((size_t)b * COUT + tid) * CTOUT + step] =
            bp[tid] + part[tid][0] + part[tid][1] + part[tid][2];
}

// ---------------------------------------------------------------------------
extern "C" void kernel_launch(void* const* d_in, const int* in_sizes, int n_in,
                              void* d_out, int out_size, void* d_ws, size_t ws_size,
                              hipStream_t stream)
{
    (void)in_sizes; (void)n_in; (void)out_size; (void)ws_size;
    const int*   spkr     = (const int*)d_in[1];
    const float* memory   = (const float*)d_in[2];
    const unsigned char* mask = (const unsigned char*)d_in[3];
    const float* spkr_emb = (const float*)d_in[4];
    const float* Wq       = (const float*)d_in[5];
    const float* Wm       = (const float*)d_in[6];
    const float* Wv       = (const float*)d_in[7];
    const float* conv_w   = (const float*)d_in[8];
    const float* loc_w    = (const float*)d_in[9];
    const float* W_ih     = (const float*)d_in[10];
    const float* W_hh     = (const float*)d_in[11];
    const float* b_ih     = (const float*)d_in[12];
    const float* b_hh     = (const float*)d_in[13];
    const float* Wp       = (const float*)d_in[14];
    const float* bp       = (const float*)d_in[15];
    float* out = (float*)d_out;

    // ---- workspace carve-up (256B aligned slices) ----
    char* ws = (char*)d_ws;
    size_t off = 0;
    auto carve = [&](size_t bytes) -> void* {
        void* p = ws + off;
        off = (off + bytes + 255) & ~(size_t)255;
        return p;
    };
    bf16_t* wcat_bf = (bf16_t*)carve((size_t)4 * CRNN * CKCAT * 2);
    bf16_t* wq_bf   = (bf16_t*)carve((size_t)CATT * CRNN * 2);
    bf16_t* wm_bf   = (bf16_t*)carve((size_t)CATT * CD * 2);
    bf16_t* mem_bf  = (bf16_t*)carve((size_t)CB * CT * CD * 2);
    float*  pm      = (float*)carve((size_t)CB * CT * CATT * 4);
    float*  bsum    = (float*)carve((size_t)4 * CRNN * 4);
    float*  hbuf    = (float*)carve((size_t)CB * CRNN * 4);
    float*  cbuf    = (float*)carve((size_t)CB * CRNN * 4);
    bf16_t* xcat    = (bf16_t*)carve((size_t)2 * CB * CKCAT * 2);  // ping-pong
    float*  aprev   = (float*)carve((size_t)CB * CT * 4);
    float*  acum    = (float*)carve((size_t)CB * CT * 4);
    float*  ebuf    = (float*)carve((size_t)CB * CT * 4);
    float*  pq      = (float*)carve((size_t)CB * CATT * 4);

    // ---- one-time: bf16 conversions, biases, processed memory ----
    build_wcat<<<(4 * CRNN * CKCAT + 255) / 256, 256, 0, stream>>>(W_ih, W_hh, wcat_bf);
    build_bsum<<<(4 * CRNN + 255) / 256, 256, 0, stream>>>(b_ih, b_hh, bsum);
    f32_to_bf16<<<(CATT * CRNN + 255) / 256, 256, 0, stream>>>(Wq, wq_bf, CATT * CRNN);
    f32_to_bf16<<<(CATT * CD + 255) / 256, 256, 0, stream>>>(Wm, wm_bf, CATT * CD);
    f32_to_bf16<<<(CB * CT * CD + 255) / 256, 256, 0, stream>>>(memory, mem_bf, CB * CT * CD);

    {   // pm[b*T, a] = memory . Wm^T   (M=32768, N=128, K=512)
        int tilesN = CATT / 16;
        int total  = (CB * CT / 16) * tilesN;
        gemm_bf16_wmma<<<(total * 32 + 255) / 256, 256, 0, stream>>>(
            mem_bf, CD, wm_bf, CD, pm, CATT, tilesN, total, CD);
    }

    // ---- state init ----
    hipMemsetAsync(hbuf, 0, (size_t)CB * CRNN * 4, stream);
    hipMemsetAsync(cbuf, 0, (size_t)CB * CRNN * 4, stream);
    hipMemsetAsync(aprev, 0, (size_t)CB * CT * 4, stream);
    hipMemsetAsync(acum, 0, (size_t)CB * CT * 4, stream);
    init_xcat<<<(2 * CB * CKCAT + 255) / 256, 256, 0, stream>>>(spkr, spkr_emb, xcat);

    // ---- recurrent decode: 4 kernels per step ----
    for (int step = 0; step < CTOUT; ++step) {
        bf16_t* xc = xcat + (size_t)(step & 1) * CB * CKCAT;
        bf16_t* xn = xcat + (size_t)((step & 1) ^ 1) * CB * CKCAT;

        // gates GEMM + LSTM cell (writes c, h, h->xn)
        gates_lstm_wmma<<<64, 256, 0, stream>>>(xc, wcat_bf, bsum, cbuf, hbuf, xn);

        {   // pq = h . Wq^T   (M=64, N=128, K=1024); h lives inside xn
            int tilesN = CATT / 16;
            int total  = (CB / 16) * tilesN;
            gemm_bf16_wmma<<<(total * 32 + 255) / 256, 256, 0, stream>>>(
                xn + (CD + CSPK), CKCAT, wq_bf, CRNN, pq, CATT, tilesN, total, CRNN);
        }

        // location conv + attention energies (reads prev-step aprev/acum)
        energies_fused<<<CB * (CT / 256), 256, 0, stream>>>(
            aprev, acum, conv_w, loc_w, pq, pm, Wv, mask, ebuf);

        // softmax + alignment update + context (-> xn) + output projection
        softmax_ctx_proj<<<CB, 256, 0, stream>>>(
            ebuf, mem_bf, hbuf, Wp, bp, aprev, acum, xn, out, step);
    }
}
